// OurAdapterLayer_71743133712481
// MI455X (gfx1250) — matrix operand order repaired and checked by reading, not compiled, not run on hardware
//
#include <hip/hip_runtime.h>

typedef __attribute__((ext_vector_type(16))) _Float16 v16h;
typedef __attribute__((ext_vector_type(8)))  _Float16 v8h;
typedef __attribute__((ext_vector_type(4)))  _Float16 v4h;
typedef __attribute__((ext_vector_type(8)))  float    v8f;
typedef __attribute__((ext_vector_type(4)))  float    v4f;

#define E_  8
#define D_  1024
#define DO_ 1024
#define R_  64
#define H_  256
#define B_  8
#define T_  2048
#define BT_ (B_*T_)

// Block tile: 128 (M) x (NI*32) (N), K-step 32. 256 threads = 8 waves in a
// 4 (M) x 2 (N) grid; each wave owns a 32 x (NI*16) sub-tile = 2 x NI WMMA
// 16x16 accumulators. NI=4 -> 128-wide block tile, NI=2 -> 64-wide.
#define BM 128
#define BK 32
#define SA 40   // LDS row stride (halves), padded vs bank conflicts
#define SB 40

// ---------------------------------------------------------------- conversions

__global__ void convert_f32_f16(const float* __restrict__ in,
                                _Float16* __restrict__ out, int n4) {
  int i = blockIdx.x * blockDim.x + threadIdx.x;
  if (i < n4) {
    v4f v = *(const v4f*)(in + (size_t)i * 4);
    v4h h;
    h.x = (_Float16)v.x; h.y = (_Float16)v.y;
    h.z = (_Float16)v.z; h.w = (_Float16)v.w;
    *(v4h*)(out + (size_t)i * 4) = h;
  }
}

// in: [E][Rr][Cc] f32 row-major  ->  out: [E][Cc][Rr] f16 (transposed)
__global__ void transpose_f32_f16(const float* __restrict__ in,
                                  _Float16* __restrict__ out, int Rr, int Cc) {
  __shared__ float tile[32][33];
  int e = blockIdx.z;
  const float* pin  = in  + (size_t)e * Rr * Cc;
  _Float16*    pout = out + (size_t)e * Rr * Cc;
  int c0 = blockIdx.x * 32, r0 = blockIdx.y * 32;
  int tx = threadIdx.x, ty = threadIdx.y;   // blockDim = (32, 8)
  #pragma unroll
  for (int j = 0; j < 32; j += 8)
    tile[ty + j][tx] = pin[(size_t)(r0 + ty + j) * Cc + c0 + tx];
  __syncthreads();
  #pragma unroll
  for (int j = 0; j < 32; j += 8)
    pout[(size_t)(c0 + ty + j) * Rr + r0 + tx] = (_Float16)tile[tx][ty + j];
}

__global__ void zero_f32(float* __restrict__ p, int n) {
  int i = blockIdx.x * blockDim.x + threadIdx.x;
  if (i < n) p[i] = 0.0f;
}

// ------------------------------------------------------------- WMMA GEMM core
// Accumulates C += A[m0.., :K] * B^T[n0.., :K]^T into acc[2][NI].
// A: f16 row-major [M, lda]; Bt: f16 row-major [N, ldb] (B transposed).
// Double-buffered LDS pipeline, one barrier per K-step.
template<int NI>
__device__ __forceinline__ void gemm_acc(const _Float16* __restrict__ A, int lda,
                                         const _Float16* __restrict__ Bt, int ldb,
                                         int K, int m0, int n0,
                                         _Float16* As, _Float16* Bs,
                                         v8f acc[2][NI]) {
  constexpr int BN  = NI * 32;       // block tile N width
  constexpr int NB  = NI / 2;        // B 16B-chunks staged per thread
  constexpr int ABUF = BM * SA;      // halves per A buffer
  constexpr int BBUF = BN * SB;      // halves per B buffer

  const int tid  = threadIdx.x;
  const int lane = tid & 31;
  const int w    = tid >> 5;
  const int wm   = w & 3;            // wave M position (0..3)
  const int wn   = w >> 2;           // wave N position (0..1)
  const int nl   = lane & 15;
  const int hi   = lane >> 4;        // half-wave select

  // Per-thread staging addresses (A: 512 chunks / 2 per thread; B: NB/thread)
  const _Float16* gA[2];
  int lA[2];
  #pragma unroll
  for (int i = 0; i < 2; i++) {
    int c = i * 256 + tid;
    int row = c >> 2, kc = c & 3;
    gA[i] = A + (size_t)(m0 + row) * lda + kc * 8;
    lA[i] = row * SA + kc * 8;
  }
  const _Float16* gB[NB];
  int lB[NB];
  #pragma unroll
  for (int j = 0; j < NB; j++) {
    int c = j * 256 + tid;
    int n = c >> 2, kc = c & 3;
    gB[j] = Bt + (size_t)(n0 + n) * ldb + kc * 8;
    lB[j] = n * SB + kc * 8;
  }

  // Prologue: stage K-tile 0 into buffer 0
  v8h ra[2], rb[NB];
  #pragma unroll
  for (int i = 0; i < 2; i++)  ra[i] = *(const v8h*)gA[i];
  #pragma unroll
  for (int j = 0; j < NB; j++) rb[j] = *(const v8h*)gB[j];
  #pragma unroll
  for (int i = 0; i < 2; i++)  *(v8h*)&As[lA[i]] = ra[i];
  #pragma unroll
  for (int j = 0; j < NB; j++) *(v8h*)&Bs[lB[j]] = rb[j];
  __syncthreads();

  int buf = 0;
  for (int kk = 0; kk < K; kk += BK) {
    const bool more = (kk + BK) < K;
    // ---- issue next tile's global loads (latency hides under WMMA)
    if (more) {
      #pragma unroll
      for (int i = 0; i < 2; i++)  ra[i] = *(const v8h*)(gA[i] + kk + BK);
      #pragma unroll
      for (int j = 0; j < NB; j++) rb[j] = *(const v8h*)(gB[j] + kk + BK);
    }
    if (kk + 2 * BK < K) {
      __builtin_prefetch(gA[0] + kk + 2 * BK, 0, 1);
      __builtin_prefetch(gA[1] + kk + 2 * BK, 0, 1);
    }

    // ---- fragments from current buffer
    // A (ISA 16-bit A 16x32): lanes 0-15 K 0..7 / 16..23; lanes 16-31 K 8..15 / 24..31
    const _Float16* Ab = As + buf * ABUF;
    const _Float16* Bb = Bs + buf * BBUF;
    v16h a[2], b[NI];
    #pragma unroll
    for (int mi = 0; mi < 2; mi++) {
      int m = wm * 32 + mi * 16 + nl;
      v8h lo = *(const v8h*)&Ab[m * SA + hi * 8];
      v8h up = *(const v8h*)&Ab[m * SA + 16 + hi * 8];
      #pragma unroll
      for (int q = 0; q < 8; q++) { a[mi][q] = lo[q]; a[mi][8 + q] = up[q]; }
    }
    // B: lane = column, lanes 0-15 K 0..7 / 16..23; lanes 16-31 K 8..15 / 24..31
    #pragma unroll
    for (int ni = 0; ni < NI; ni++) {
      int n = wn * (NI * 16) + ni * 16 + nl;
      v8h lo = *(const v8h*)&Bb[n * SB + hi * 8];
      v8h up = *(const v8h*)&Bb[n * SB + 16 + hi * 8];
      #pragma unroll
      for (int q = 0; q < 8; q++) { b[ni][q] = lo[q]; b[ni][8 + q] = up[q]; }
    }

    #pragma unroll
    for (int mi = 0; mi < 2; mi++)
      #pragma unroll
      for (int ni = 0; ni < NI; ni++)
        acc[mi][ni] = __builtin_amdgcn_wmma_f32_16x16x32_f16(
            false, a[mi], false, b[ni], (short)0, acc[mi][ni], false, false);

    // ---- commit staged registers into the other buffer
    if (more) {
      _Float16* An = As + (buf ^ 1) * ABUF;
      _Float16* Bn = Bs + (buf ^ 1) * BBUF;
      #pragma unroll
      for (int i = 0; i < 2; i++)  *(v8h*)&An[lA[i]] = ra[i];
      #pragma unroll
      for (int j = 0; j < NB; j++) *(v8h*)&Bn[lB[j]] = rb[j];
    }
    __syncthreads();
    buf ^= 1;
  }
}

#define EPILOGUE_COORDS                                            \
  const int tid = threadIdx.x;                                     \
  const int lane = tid & 31, w = tid >> 5;                         \
  const int wm = w & 3, wn = w >> 2;                               \
  const int nl = lane & 15, hi = lane >> 4;

// ------------------------------------------------------------- stage kernels

// h = relu(x * Wd1[e] + bd1[e])   -> hall[e] (f16 [BT, H])
template<int NI>
__global__ __launch_bounds__(256) void disc1_kernel(
    const _Float16* __restrict__ xh, const _Float16* __restrict__ Wd1T,
    const float* __restrict__ bd1, _Float16* __restrict__ hall) {
  __shared__ _Float16 As[2 * BM * SA];
  __shared__ _Float16 Bs[2 * NI * 32 * SB];
  int e = blockIdx.z;
  int m0 = blockIdx.y * BM, n0 = blockIdx.x * (NI * 32);
  v8f acc[2][NI] = {};
  gemm_acc<NI>(xh, D_, Wd1T + (size_t)e * H_ * D_, D_, D_, m0, n0, As, Bs, acc);
  EPILOGUE_COORDS
  const float* bias = bd1 + e * H_;
  _Float16* out = hall + (size_t)e * BT_ * H_;
  #pragma unroll
  for (int mi = 0; mi < 2; mi++)
    #pragma unroll
    for (int ni = 0; ni < NI; ni++) {
      int col = n0 + wn * (NI * 16) + ni * 16 + nl;
      float bb = bias[col];
      #pragma unroll
      for (int r = 0; r < 8; r++) {
        int row = m0 + wm * 32 + mi * 16 + hi * 8 + r;
        float v = acc[mi][ni][r] + bb;
        out[(size_t)row * H_ + col] = (_Float16)(v > 0.0f ? v : 0.0f);
      }
    }
}

// recon = h * Wd2[e] + bd2[e]; losses[e,b] += sum((recon - x)^2)
template<int NI>
__global__ __launch_bounds__(256) void disc2_loss_kernel(
    const _Float16* __restrict__ hall, const _Float16* __restrict__ Wd2T,
    const float* __restrict__ bd2, const float* __restrict__ x,
    float* __restrict__ losses) {
  __shared__ _Float16 As[2 * BM * SA];
  __shared__ _Float16 Bs[2 * NI * 32 * SB];
  __shared__ float red[256];
  int e = blockIdx.z;
  int m0 = blockIdx.y * BM, n0 = blockIdx.x * (NI * 32);
  v8f acc[2][NI] = {};
  gemm_acc<NI>(hall + (size_t)e * BT_ * H_, H_, Wd2T + (size_t)e * D_ * H_, H_,
               H_, m0, n0, As, Bs, acc);
  EPILOGUE_COORDS
  const float* bias = bd2 + e * D_;
  float local = 0.0f;
  #pragma unroll
  for (int mi = 0; mi < 2; mi++)
    #pragma unroll
    for (int ni = 0; ni < NI; ni++) {
      int col = n0 + wn * (NI * 16) + ni * 16 + nl;
      float bb = bias[col];
      #pragma unroll
      for (int r = 0; r < 8; r++) {
        int row = m0 + wm * 32 + mi * 16 + hi * 8 + r;
        float d = acc[mi][ni][r] + bb - x[(size_t)row * D_ + col];
        local += d * d;
      }
    }
  red[tid] = local;
  __syncthreads();
  for (int s = 128; s > 0; s >>= 1) {
    if (tid < s) red[tid] += red[tid + s];
    __syncthreads();
  }
  if (tid == 0) atomicAdd(&losses[e * B_ + (m0 / T_)], red[0]);
}

__global__ void argmin_kernel(const float* __restrict__ losses,
                              int* __restrict__ idx) {
  int b = threadIdx.x;
  if (b < B_) {
    float best = losses[b];
    int bi = 0;
    for (int e = 1; e < E_; e++) {
      float v = losses[e * B_ + b];
      if (v < best) { best = v; bi = e; }
    }
    idx[b] = bi;
  }
}

// a = relu(x * Wa_down[idx[b]] + ba_down[idx[b]])  -> ah (f16 [BT, R])
template<int NI>
__global__ __launch_bounds__(256) void adapter_down_kernel(
    const _Float16* __restrict__ xh, const _Float16* __restrict__ WadT,
    const float* __restrict__ bad, const int* __restrict__ idxb,
    _Float16* __restrict__ ah) {
  __shared__ _Float16 As[2 * BM * SA];
  __shared__ _Float16 Bs[2 * NI * 32 * SB];
  int m0 = blockIdx.y * BM, n0 = blockIdx.x * (NI * 32);
  int e = idxb[m0 / T_];
  v8f acc[2][NI] = {};
  gemm_acc<NI>(xh, D_, WadT + (size_t)e * R_ * D_, D_, D_, m0, n0, As, Bs, acc);
  EPILOGUE_COORDS
  const float* bias = bad + e * R_;
  #pragma unroll
  for (int mi = 0; mi < 2; mi++)
    #pragma unroll
    for (int ni = 0; ni < NI; ni++) {
      int col = n0 + wn * (NI * 16) + ni * 16 + nl;
      float bb = bias[col];
      #pragma unroll
      for (int r = 0; r < 8; r++) {
        int row = m0 + wm * 32 + mi * 16 + hi * 8 + r;
        float v = acc[mi][ni][r] + bb;
        ah[(size_t)row * R_ + col] = (_Float16)(v > 0.0f ? v : 0.0f);
      }
    }
}

// out = x*W_base + b_base + a*Wa_up[idx[b]] + ba_up[idx[b]]   (f32)
template<int NI>
__global__ __launch_bounds__(256) void final_kernel(
    const _Float16* __restrict__ xh, const _Float16* __restrict__ WbT,
    const float* __restrict__ b_base, const _Float16* __restrict__ ah,
    const _Float16* __restrict__ WauT, const float* __restrict__ bau,
    const int* __restrict__ idxb, float* __restrict__ out) {
  __shared__ _Float16 As[2 * BM * SA];
  __shared__ _Float16 Bs[2 * NI * 32 * SB];
  int m0 = blockIdx.y * BM, n0 = blockIdx.x * (NI * 32);
  int e = idxb[m0 / T_];
  v8f acc[2][NI] = {};
  gemm_acc<NI>(xh, D_, WbT, D_, D_, m0, n0, As, Bs, acc);                  // base
  gemm_acc<NI>(ah, R_, WauT + (size_t)e * DO_ * R_, R_, R_, m0, n0, As, Bs,
               acc);                                                       // adapter up
  EPILOGUE_COORDS
  const float* bias_e = bau + e * DO_;
  #pragma unroll
  for (int mi = 0; mi < 2; mi++)
    #pragma unroll
    for (int ni = 0; ni < NI; ni++) {
      int col = n0 + wn * (NI * 16) + ni * 16 + nl;
      float bb = b_base[col] + bias_e[col];
      #pragma unroll
      for (int r = 0; r < 8; r++) {
        int row = m0 + wm * 32 + mi * 16 + hi * 8 + r;
        out[(size_t)row * DO_ + col] = acc[mi][ni][r] + bb;
      }
    }
}

// ----------------------------------------------------------------- launcher

extern "C" void kernel_launch(void* const* d_in, const int* in_sizes, int n_in,
                              void* d_out, int out_size, void* d_ws, size_t ws_size,
                              hipStream_t stream) {
  (void)in_sizes; (void)n_in; (void)out_size; (void)ws_size;
  const float* x      = (const float*)d_in[0];
  const float* W_base = (const float*)d_in[1];
  const float* b_base = (const float*)d_in[2];
  const float* Wd1    = (const float*)d_in[3];
  const float* bd1    = (const float*)d_in[4];
  const float* Wd2    = (const float*)d_in[5];
  const float* bd2    = (const float*)d_in[6];
  const float* Wad    = (const float*)d_in[7];
  const float* bad    = (const float*)d_in[8];
  const float* Wau    = (const float*)d_in[9];
  const float* bau    = (const float*)d_in[10];
  float* out = (float*)d_out;

  char* ws = (char*)d_ws;
  size_t off = 0;
  auto alloc = [&](size_t bytes) {
    char* p = ws + off;
    off += (bytes + 255) & ~(size_t)255;
    return p;
  };
  _Float16* xh   = (_Float16*)alloc((size_t)BT_ * D_ * 2);       // 32 MB
  _Float16* WbT  = (_Float16*)alloc((size_t)D_ * DO_ * 2);       //  2 MB
  _Float16* Wd1T = (_Float16*)alloc((size_t)E_ * D_ * H_ * 2);   //  4 MB
  _Float16* Wd2T = (_Float16*)alloc((size_t)E_ * H_ * D_ * 2);   //  4 MB
  _Float16* WadT = (_Float16*)alloc((size_t)E_ * D_ * R_ * 2);   //  1 MB
  _Float16* WauT = (_Float16*)alloc((size_t)E_ * R_ * DO_ * 2);  //  1 MB
  _Float16* hall = (_Float16*)alloc((size_t)E_ * BT_ * H_ * 2);  // 64 MB
  _Float16* ah   = (_Float16*)alloc((size_t)BT_ * R_ * 2);       //  2 MB
  float*    loss = (float*)   alloc(E_ * B_ * 4);
  int*      idxb = (int*)     alloc(B_ * 4);

  // ---- precision/layout conversion (f32 -> f16, weights transposed N-major)
  convert_f32_f16<<<(BT_ * D_ / 4 + 255) / 256, 256, 0, stream>>>(x, xh, BT_ * D_ / 4);
  transpose_f32_f16<<<dim3(DO_/32, D_/32, 1 ), dim3(32, 8), 0, stream>>>(W_base, WbT,  D_, DO_);
  transpose_f32_f16<<<dim3(H_ /32, D_/32, E_), dim3(32, 8), 0, stream>>>(Wd1,    Wd1T, D_, H_ );
  transpose_f32_f16<<<dim3(D_ /32, H_/32, E_), dim3(32, 8), 0, stream>>>(Wd2,    Wd2T, H_, D_ );
  transpose_f32_f16<<<dim3(R_ /32, D_/32, E_), dim3(32, 8), 0, stream>>>(Wad,    WadT, D_, R_ );
  transpose_f32_f16<<<dim3(DO_/32, R_/32, E_), dim3(32, 8), 0, stream>>>(Wau,    WauT, R_, DO_);
  zero_f32<<<1, 64, 0, stream>>>(loss, E_ * B_);

  // ---- discriminators (bulk of the FLOPs) + routing
  disc1_kernel<4><<<dim3(H_ / 128, BT_ / BM, E_), 256, 0, stream>>>(xh, Wd1T, bd1, hall);
  disc2_loss_kernel<4><<<dim3(D_ / 128, BT_ / BM, E_), 256, 0, stream>>>(hall, Wd2T, bd2, x, loss);
  argmin_kernel<<<1, 32, 0, stream>>>(loss, idxb);

  // ---- routed adapter + fused base GEMM
  adapter_down_kernel<2><<<dim3(R_ / 64, BT_ / BM, 1), 256, 0, stream>>>(xh, WadT, bad, idxb, ah);
  final_kernel<4><<<dim3(DO_ / 128, BT_ / BM, 1), 256, 0, stream>>>(xh, WbT, b_base, ah, WauT, bau, idxb, out);
}
